// ATNAggregation2d_37091337568322
// MI455X (gfx1250) — matrix operand order repaired and checked by reading, not compiled, not run on hardware
//
#include <hip/hip_runtime.h>

// MI455X / gfx1250: wave32, WMMA 16x16x32 f16 -> f32 accum.
typedef __attribute__((ext_vector_type(16))) _Float16 v16h;
typedef __attribute__((ext_vector_type(8)))  float    v8f;

#define MDIM 8
#define BDIM 2
#define CDIM 64
#define HW   9216   // 96*96

// ---------------------------------------------------------------------------
// Kernel 0 (one-shot, tiny): fold the chained linears.
//   G_i = W_i @ C     (64x64)           out = X @ G_i^T + g_b
//   g_b = W_i @ c_b + w_b
// ws layout (floats): [0..4095]=Gk  [4096..8191]=Ga  [8192..12287]=Gq
//                     [12288..12479]=fused biases (K,A,Q each 64)
// ---------------------------------------------------------------------------
__global__ void fuse_weights_kernel(const float* __restrict__ c_w,
                                    const float* __restrict__ c_b,
                                    const float* __restrict__ wq_w, const float* __restrict__ wq_b,
                                    const float* __restrict__ wk_w, const float* __restrict__ wk_b,
                                    const float* __restrict__ wa_w, const float* __restrict__ wa_b,
                                    float* __restrict__ ws) {
    const int i = blockIdx.x;                      // 0=K, 1=A, 2=Q
    const float* W  = (i == 0) ? wk_w : (i == 1) ? wa_w : wq_w;
    const float* wb = (i == 0) ? wk_b : (i == 1) ? wa_b : wq_b;
    const int t = threadIdx.x;
    for (int o = t; o < 4096; o += 256) {
        const int n = o >> 6, k = o & 63;
        float g = 0.f;
#pragma unroll 8
        for (int j = 0; j < 64; ++j) g += W[n * 64 + j] * c_w[j * 64 + k];
        ws[i * 4096 + o] = g;
    }
    if (t < 64) {
        float g = wb[t];
#pragma unroll 8
        for (int j = 0; j < 64; ++j) g += W[t * 64 + j] * c_b[j];
        ws[3 * 4096 + i * 64 + t] = g;
    }
}

// ---------------------------------------------------------------------------
// Main kernel: block = 256 threads (8 waves) = 16 consecutive pixels.
// Wave w handles process m=w: loads X A-fragment (16 px x 64 ch) from global,
// runs K and A GEMMs with v_wmma_f32_16x16x32_f16; wave 0 averages X frags
// over m in LDS and runs the Q GEMM; VALU epilogue does the M=8 softmax.
// ---------------------------------------------------------------------------
__global__ __launch_bounds__(256)
void atn_attention_kernel(const float* __restrict__ xs,
                          const float* __restrict__ ws,
                          float* __restrict__ out) {
    // Weight B-fragments, f16, pre-swizzled: [mat][kchunk][ntile][lane] = 16 halves.
    __shared__ v16h  sWB[3][2][4][32];   // 24 KB
    __shared__ float sBias[3][64];       // 0.75 KB
    __shared__ v16h  sX[MDIM][2][32];    // X A-fragments per wave, 16 KB
    __shared__ float sK[MDIM][16][64];   // 32 KB
    __shared__ float sA[MDIM][16][64];   // 32 KB
    __shared__ float sQ[16][64];         // 4 KB
    __shared__ float sS[16][MDIM];       // scores, 0.5 KB

    const int t    = threadIdx.x;
    const int wave = t >> 5;
    const int lane = t & 31;
    const int hi   = lane >> 4;          // lane-half selects K sub-range
    const int r    = lane & 15;          // row (= pixel-in-tile) / col-in-tile
    const int pix0 = blockIdx.x * 16;
    const int b    = pix0 / HW;          // 16 | HW, so whole block shares b
    const int rem0 = pix0 - b * HW;

    // ---- Phase 0: stage fused weights (f32 -> f16) into B-fragment layout.
    // B fragment as staged here (32x16, f16): element j of v16h
    // holds (k = hi*16 + j, n = lane&15).  B[k][n] = G[n][k]  (out = X @ G^T).
    for (int idx = t; idx < 3 * 4096; idx += 256) {
        const int i  = idx >> 12;
        const int rm = idx & 4095;
        const int kc = rm >> 11;
        const int nt = (rm >> 9) & 3;
        const int ln = (rm >> 4) & 31;
        const int j  = rm & 15;
        const int k  = kc * 32 + (ln >> 4) * 16 + j;
        const int n  = nt * 16 + (ln & 15);
        ((_Float16*)&sWB[i][kc][nt][ln])[j] = (_Float16)ws[i * 4096 + n * 64 + k];
    }
    if (t < 192) sBias[t >> 6][t & 63] = ws[3 * 4096 + t];
    __syncthreads();

    // ---- Phase 1: load X A-fragments from global (coalesced across row-lanes).
    // A operand (16x32 f16): lane L holds row L&15; element j: k = hi*16 + j (same k order as B)
    // Channel c = kc*32 + k; channel stride in xs is HW.
    const int m = wave;
    const float* xbase = xs + (size_t)((m * BDIM + b) * CDIM) * HW + rem0 + r;
    v16h a0, a1;
#pragma unroll
    for (int kc = 0; kc < 2; ++kc) {
        v16h av;
#pragma unroll
        for (int j = 0; j < 16; ++j) {
            const int g = j >> 1;
            const int k = hi * 16 + j; (void)g;
            const int c = kc * 32 + k;
            ((_Float16*)&av)[j] = (_Float16)xbase[(size_t)c * HW];
        }
        if (kc == 0) a0 = av; else a1 = av;
        sX[wave][kc][lane] = av;           // stage for cross-m mean (Q path)
    }

    // ---- K and A GEMMs: 16 pixels x 64 ch, f32 accumulate.
#pragma unroll
    for (int i = 0; i < 2; ++i) {
        float* dst = (i == 0) ? &sK[m][0][0] : &sA[m][0][0];
#pragma unroll
        for (int nt = 0; nt < 4; ++nt) {
            v8f acc = {};
            acc = __builtin_amdgcn_wmma_f32_16x16x32_f16(
                false, a0, false, sWB[i][0][nt][lane], (short)0, acc, false, false);
            acc = __builtin_amdgcn_wmma_f32_16x16x32_f16(
                false, a1, false, sWB[i][1][nt][lane], (short)0, acc, false, false);
            const float bias = sBias[i][nt * 16 + r];
            // D layout: lane&15 = n, VGPR g = row (rows 8..15 on lane-half 1)
#pragma unroll
            for (int g = 0; g < 8; ++g)
                dst[(g + hi * 8) * 64 + nt * 16 + r] = acc[g] + bias;
        }
    }
    __syncthreads();

    // ---- Phase 2: wave 0 builds mean_m(X) fragment (packed f16 adds) + Q GEMM.
    if (wave == 0) {
        v16h q0, q1;
#pragma unroll
        for (int kc = 0; kc < 2; ++kc) {
            v16h s = sX[0][kc][lane];
#pragma unroll
            for (int w = 1; w < MDIM; ++w) s = s + sX[w][kc][lane];  // v_pk_add_f16
            s = s * (_Float16)0.125f;                                 // 1/M
            if (kc == 0) q0 = s; else q1 = s;
        }
#pragma unroll
        for (int nt = 0; nt < 4; ++nt) {
            v8f acc = {};
            acc = __builtin_amdgcn_wmma_f32_16x16x32_f16(
                false, q0, false, sWB[2][0][nt][lane], (short)0, acc, false, false);
            acc = __builtin_amdgcn_wmma_f32_16x16x32_f16(
                false, q1, false, sWB[2][1][nt][lane], (short)0, acc, false, false);
            const float bias = sBias[2][nt * 16 + r];
#pragma unroll
            for (int g = 0; g < 8; ++g)
                sQ[g + hi * 8][nt * 16 + r] = acc[g] + bias;
        }
    }
    __syncthreads();

    // ---- Phase 3a: scores[p][m] = (Q[p] . K[m][p]) / sqrt(C).  128 threads.
    if (t < 128) {
        const int mm = t >> 4, p = t & 15;
        float s = 0.f;
#pragma unroll 8
        for (int c = 0; c < 64; ++c) s += sQ[p][c] * sK[mm][p][c];
        sS[p][mm] = s * 0.125f;   // 1/sqrt(64)
    }
    __syncthreads();

    // ---- Phase 3b: softmax over M=8 + weighted sum of A, coalesced store.
    {
        const int p = t & 15, cg = t >> 4;   // 4 channels per thread
        float sc[MDIM];
        float mx = -1e30f;
#pragma unroll
        for (int mm = 0; mm < MDIM; ++mm) { sc[mm] = sS[p][mm]; mx = fmaxf(mx, sc[mm]); }
        float sum = 0.f;
#pragma unroll
        for (int mm = 0; mm < MDIM; ++mm) { sc[mm] = __expf(sc[mm] - mx); sum += sc[mm]; }
        const float inv = 1.0f / sum;
        float z[4] = {0.f, 0.f, 0.f, 0.f};
#pragma unroll
        for (int mm = 0; mm < MDIM; ++mm) {
            const float al = sc[mm] * inv;
#pragma unroll
            for (int cc = 0; cc < 4; ++cc) z[cc] += al * sA[mm][p][cg * 4 + cc];
        }
#pragma unroll
        for (int cc = 0; cc < 4; ++cc) {
            const int c = cg * 4 + cc;
            out[(size_t)(b * CDIM + c) * HW + rem0 + p] = z[cc];  // [B,C,H,W]
        }
    }
}

// ---------------------------------------------------------------------------
extern "C" void kernel_launch(void* const* d_in, const int* in_sizes, int n_in,
                              void* d_out, int out_size, void* d_ws, size_t ws_size,
                              hipStream_t stream) {
    (void)in_sizes; (void)n_in; (void)out_size; (void)ws_size;
    const float* xs   = (const float*)d_in[0];
    const float* c_w  = (const float*)d_in[1];
    const float* c_b  = (const float*)d_in[2];
    const float* wq_w = (const float*)d_in[3];
    const float* wq_b = (const float*)d_in[4];
    const float* wk_w = (const float*)d_in[5];
    const float* wk_b = (const float*)d_in[6];
    const float* wa_w = (const float*)d_in[7];
    const float* wa_b = (const float*)d_in[8];
    float* ws  = (float*)d_ws;    // needs ~50 KB scratch
    float* out = (float*)d_out;

    fuse_weights_kernel<<<3, 256, 0, stream>>>(c_w, c_b, wq_w, wq_b,
                                               wk_w, wk_b, wa_w, wa_b, ws);
    // 18432 pixels / 16 per block = 1152 blocks
    atn_attention_kernel<<<1152, 256, 0, stream>>>(xs, ws, out);
}
